// MultiHeadedAttention_24257975288113
// MI455X (gfx1250) — matrix operand order repaired and checked by reading, not compiled, not run on hardware
//
#include <hip/hip_runtime.h>

// ---------------- CDNA5 WMMA types ----------------
typedef __bf16 bf16;
typedef __attribute__((ext_vector_type(16))) __bf16 v16bf;
typedef __attribute__((ext_vector_type(8)))  float  v8f;

union FragAB { v16bf v; uint4 q[2]; };           // 32B: A or B operand (16 bf16 / lane)
union FragC  { v8f   v; float f[8]; };           // 32B: C/D accumulator (8 f32 / lane)

#define B_  4
#define S_  2048
#define E_  2048
#define H_  16
#define DH_ 128

__device__ __forceinline__ bf16 f2bf(float f) { return (bf16)f; }  // native v_cvt path

__device__ __forceinline__ v8f wmma_bf16(v16bf a, v16bf b, v8f c) {
  // D = A(16x32 bf16) x B(32x16 bf16) + C(16x16 f32)
  return __builtin_amdgcn_wmma_f32_16x16x32_bf16(false, a, false, b, (short)0, c,
                                                 false, false);
}

// =====================================================================
// Kernel 1: QKV projection.  For (b,h,which): Y = X[b] (SxE) @ W[h] (ExDH)
// Output stored bf16 as [bh][s][d].
// Grid: (S/128, B*H, 3), block 256 (8 waves). K-step = 64.
// =====================================================================
__global__ __launch_bounds__(256) void qkv_proj_kernel(
    const float* __restrict__ X,
    const float* __restrict__ Wq,
    const float* __restrict__ Wk,
    const float* __restrict__ Wv,
    bf16* __restrict__ Qo,
    bf16* __restrict__ Ko,
    bf16* __restrict__ Vo)
{
  __shared__ __align__(16) bf16 Xs[128 * 72];  // [row][e], stride 72 (64 + pad 8)
  __shared__ __align__(16) bf16 Ws[128 * 72];  // [d][e] (transposed), stride 72

  const int sblk  = blockIdx.x;        // 128-row tile of S
  const int bh    = blockIdx.y;        // b*H + h
  const int which = blockIdx.z;        // 0:Q 1:K 2:V
  const int h     = bh & (H_ - 1);

  const float* Wsel = (which == 0) ? Wq : (which == 1) ? Wk : Wv;
  bf16* Osel = (which == 0) ? Qo : (which == 1) ? Ko : Vo;

  const float* Wh = Wsel + (size_t)h * E_ * DH_;                       // [E][DH]
  const float* Xb = X + ((size_t)(bh >> 4) * S_ + (size_t)sblk * 128) * E_;
  bf16* Ob = Osel + ((size_t)bh * S_ + (size_t)sblk * 128) * DH_;

  const int tid  = threadIdx.x;
  const int wave = tid >> 5;
  const int lane = tid & 31;
  const int lo   = lane & 15;
  const int hi   = lane >> 4;
  const int wr   = wave & 3;           // 32-row group
  const int wc   = wave >> 2;          // 64-col group

  FragC acc[2][4];
  for (int m = 0; m < 2; ++m)
    for (int n = 0; n < 4; ++n)
      for (int r = 0; r < 8; ++r) acc[m][n].f[r] = 0.0f;

  for (int e0 = 0; e0 < E_; e0 += 64) {
    __syncthreads();
    // ---- stage X tile (128 x 64) f32 -> bf16 LDS ----
#pragma unroll
    for (int i = 0; i < 8; ++i) {
      int idx = (tid + i * 256) * 4;                 // 0..8191
      int r = idx >> 6, c = idx & 63;
      float4 fv = *(const float4*)(Xb + (size_t)r * E_ + e0 + c);
      bf16* p = &Xs[r * 72 + c];
      p[0] = f2bf(fv.x); p[1] = f2bf(fv.y); p[2] = f2bf(fv.z); p[3] = f2bf(fv.w);
    }
    // ---- stage W tile (64 x 128) transposed -> Ws[d][e] ----
#pragma unroll
    for (int i = 0; i < 8; ++i) {
      int idx = (tid + i * 256) * 4;                 // 0..8191 over (e,d)
      int e = idx >> 7, d = idx & 127;
      float4 fv = *(const float4*)(Wh + (size_t)(e0 + e) * DH_ + d);
      Ws[(d + 0) * 72 + e] = f2bf(fv.x);
      Ws[(d + 1) * 72 + e] = f2bf(fv.y);
      Ws[(d + 2) * 72 + e] = f2bf(fv.z);
      Ws[(d + 3) * 72 + e] = f2bf(fv.w);
    }
    __syncthreads();

    // ---- fragments + 16 WMMAs per wave per k-step ----
    FragAB a[2][2], bfr[2][4];
#pragma unroll
    for (int kc = 0; kc < 2; ++kc) {
#pragma unroll
      for (int m = 0; m < 2; ++m) {
        int row = wr * 32 + m * 16 + lo;
        a[kc][m].q[0] = *(const uint4*)&Xs[row * 72 + kc * 32 + hi * 8];
        a[kc][m].q[1] = *(const uint4*)&Xs[row * 72 + kc * 32 + 16 + hi * 8];
      }
#pragma unroll
      for (int n = 0; n < 4; ++n) {
        int d = wc * 64 + n * 16 + lo;
        bfr[kc][n].q[0] = *(const uint4*)&Ws[d * 72 + kc * 32 + hi * 8];
        bfr[kc][n].q[1] = *(const uint4*)&Ws[d * 72 + kc * 32 + 16 + hi * 8];
      }
    }
#pragma unroll
    for (int kc = 0; kc < 2; ++kc)
#pragma unroll
      for (int m = 0; m < 2; ++m)
#pragma unroll
        for (int n = 0; n < 4; ++n)
          acc[m][n].v = wmma_bf16(a[kc][m].v, bfr[kc][n].v, acc[m][n].v);
  }

  // ---- store bf16 result: C-layout row = r + 8*hi, col = lane&15 ----
#pragma unroll
  for (int m = 0; m < 2; ++m)
#pragma unroll
    for (int n = 0; n < 4; ++n)
#pragma unroll
      for (int r = 0; r < 8; ++r) {
        int srow = wr * 32 + m * 16 + r + 8 * hi;
        int d    = wc * 64 + n * 16 + lo;
        Ob[(size_t)srow * DH_ + d] = f2bf(acc[m][n].f[r]);
      }
}

// =====================================================================
// Kernel 2: causal flash attention over bf16 Q/K/V, f32 accumulate.
// Grid: (S/64, B*H), block 128 (4 waves). Wave w owns 16 Q rows.
// =====================================================================
__global__ __launch_bounds__(128) void flash_attn_kernel(
    const bf16* __restrict__ Qb,
    const bf16* __restrict__ Kb,
    const bf16* __restrict__ Vb,
    float* __restrict__ out)
{
  __shared__ __align__(16) bf16 Ks[64 * 136];   // K block [t][d], pad stride
  __shared__ __align__(16) bf16 Vt[128 * 72];   // V block transposed [d][t]
  __shared__ __align__(16) bf16 Ps[4][16 * 72]; // per-wave P bounce [m][t]

  const int qblk = blockIdx.x;
  const int bh   = blockIdx.y;
  const int b    = bh >> 4;
  const int h    = bh & 15;

  const int tid  = threadIdx.x;
  const int wave = tid >> 5;
  const int lane = tid & 31;
  const int lo   = lane & 15;
  const int hi   = lane >> 4;

  const bf16* Qh = Qb + (size_t)bh * S_ * DH_;
  const bf16* Kh = Kb + (size_t)bh * S_ * DH_;
  const bf16* Vh = Vb + (size_t)bh * S_ * DH_;

  const int q0 = qblk * 64 + wave * 16;   // this wave's first Q row

  // ---- Q rows as 4 A-fragments (16 x 128 bf16), loaded straight from global ----
  FragAB aq[4];
#pragma unroll
  for (int c = 0; c < 4; ++c) {
    const bf16* qp = Qh + (size_t)(q0 + lo) * DH_ + c * 32 + hi * 8;
    aq[c].q[0] = *(const uint4*)qp;
    aq[c].q[1] = *(const uint4*)(qp + 16);
  }

  float row_max[8], row_sum[8];
  FragC o[8];
#pragma unroll
  for (int r = 0; r < 8; ++r) { row_max[r] = -1e30f; row_sum[r] = 0.0f; }
#pragma unroll
  for (int n = 0; n < 8; ++n)
#pragma unroll
    for (int r = 0; r < 8; ++r) o[n].f[r] = 0.0f;

  const float scale = 0.088388347648318447f;  // 1/sqrt(128)

  for (int kb = 0; kb <= qblk; ++kb) {
    __syncthreads();
    // ---- stage K block (64 x 128) row-major ----
#pragma unroll
    for (int i = 0; i < 8; ++i) {
      int idx = (tid + i * 128) * 8;
      int t = idx >> 7, d = idx & 127;
      *(uint4*)&Ks[t * 136 + d] = *(const uint4*)(Kh + (size_t)(kb * 64 + t) * DH_ + d);
    }
    // ---- stage V block transposed -> Vt[d][t] ----
#pragma unroll
    for (int i = 0; i < 8; ++i) {
      int idx = (tid + i * 128) * 8;
      int t = idx >> 7, d = idx & 127;
      union { uint4 q; bf16 u[8]; } tmp;
      tmp.q = *(const uint4*)(Vh + (size_t)(kb * 64 + t) * DH_ + d);
#pragma unroll
      for (int j = 0; j < 8; ++j) Vt[(d + j) * 72 + t] = tmp.u[j];
    }
    if (kb < qblk) {
      // warm L2 with next K block (global_prefetch_b8)
      __builtin_prefetch(Kh + (size_t)((kb + 1) * 64 + (tid >> 1)) * DH_, 0, 1);
    }
    __syncthreads();

    // ---- S = Q K^T : 4 n-tiles x 4 k-steps of WMMA ----
    FragC sc[4];
#pragma unroll
    for (int nt = 0; nt < 4; ++nt) {
      FragC cfrag;
#pragma unroll
      for (int r = 0; r < 8; ++r) cfrag.f[r] = 0.0f;
#pragma unroll
      for (int kc = 0; kc < 4; ++kc) {
        FragAB bk;
        const bf16* kp = &Ks[(nt * 16 + lo) * 136 + kc * 32 + hi * 8];
        bk.q[0] = *(const uint4*)kp;
        bk.q[1] = *(const uint4*)(kp + 16);
        cfrag.v = wmma_bf16(aq[kc].v, bk.v, cfrag.v);
      }
      sc[nt] = cfrag;
    }

    // ---- scale + causal mask (only the diagonal block needs masking) ----
    const bool diag = (kb == qblk);
#pragma unroll
    for (int nt = 0; nt < 4; ++nt)
#pragma unroll
      for (int r = 0; r < 8; ++r) {
        float v = sc[nt].f[r] * scale;
        if (diag) {
          int srow = q0 + r + 8 * hi;
          int tcol = kb * 64 + nt * 16 + lo;
          if (tcol > srow) v = -1e30f;
        }
        sc[nt].f[r] = v;
      }

    // ---- block row-max (butterfly across 16-lane half, wave32) ----
    float bm[8];
#pragma unroll
    for (int r = 0; r < 8; ++r) {
      float m = fmaxf(fmaxf(sc[0].f[r], sc[1].f[r]), fmaxf(sc[2].f[r], sc[3].f[r]));
#pragma unroll
      for (int x = 1; x < 16; x <<= 1) m = fmaxf(m, __shfl_xor(m, x, 32));
      bm[r] = m;
    }

    float alpha[8];
#pragma unroll
    for (int r = 0; r < 8; ++r) {
      float mnew = fmaxf(row_max[r], bm[r]);
      alpha[r] = __expf(row_max[r] - mnew);
      row_max[r] = mnew;
    }

    // ---- P = exp(s - m), block row-sum ----
    float bs[8];
#pragma unroll
    for (int r = 0; r < 8; ++r) bs[r] = 0.0f;
#pragma unroll
    for (int nt = 0; nt < 4; ++nt)
#pragma unroll
      for (int r = 0; r < 8; ++r) {
        float p = __expf(sc[nt].f[r] - row_max[r]);
        sc[nt].f[r] = p;
        bs[r] += p;
      }
#pragma unroll
    for (int r = 0; r < 8; ++r) {
      float s = bs[r];
#pragma unroll
      for (int x = 1; x < 16; x <<= 1) s += __shfl_xor(s, x, 32);
      row_sum[r] = row_sum[r] * alpha[r] + s;
    }

    // ---- rescale running output ----
#pragma unroll
    for (int n = 0; n < 8; ++n)
#pragma unroll
      for (int r = 0; r < 8; ++r) o[n].f[r] *= alpha[r];

    // ---- P (C-layout) -> per-wave LDS -> A-fragments ----
    bf16* pw = &Ps[wave][0];
#pragma unroll
    for (int nt = 0; nt < 4; ++nt)
#pragma unroll
      for (int r = 0; r < 8; ++r)
        pw[(r + 8 * hi) * 72 + nt * 16 + lo] = f2bf(sc[nt].f[r]);

    // ---- O += P V : 2 k-steps x 8 d-tiles ----
#pragma unroll
    for (int kc = 0; kc < 2; ++kc) {
      FragAB pa;
      const bf16* pp = &pw[lo * 72 + kc * 32 + hi * 8];
      pa.q[0] = *(const uint4*)pp;
      pa.q[1] = *(const uint4*)(pp + 16);
#pragma unroll
      for (int nt = 0; nt < 8; ++nt) {
        FragAB bv;
        const bf16* vp = &Vt[(nt * 16 + lo) * 72 + kc * 32 + hi * 8];
        bv.q[0] = *(const uint4*)vp;
        bv.q[1] = *(const uint4*)(vp + 16);
        o[nt].v = wmma_bf16(pa.v, bv.v, o[nt].v);
      }
    }
  }

  // ---- epilogue: normalize, round(.,4), head-concat layout [B][S][H*DH] ----
  float* op = out + (size_t)b * S_ * (H_ * DH_) + h * DH_;
#pragma unroll
  for (int nt = 0; nt < 8; ++nt) {
#pragma unroll
    for (int r = 0; r < 8; ++r) {
      int srow = q0 + r + 8 * hi;
      int d = nt * 16 + lo;
      float inv = __builtin_amdgcn_rcpf(row_sum[r]);
      float v = o[nt].f[r] * inv;
      v = rintf(v * 1e4f) * 1e-4f;
      op[(size_t)srow * (H_ * DH_) + d] = v;
    }
  }
}

// =====================================================================
extern "C" void kernel_launch(void* const* d_in, const int* in_sizes, int n_in,
                              void* d_out, int out_size, void* d_ws, size_t ws_size,
                              hipStream_t stream) {
  (void)in_sizes; (void)n_in; (void)out_size; (void)ws_size;

  const float* X  = (const float*)d_in[0];
  const float* Wq = (const float*)d_in[1];
  const float* Wk = (const float*)d_in[2];
  const float* Wv = (const float*)d_in[3];

  const size_t perMat = (size_t)B_ * H_ * S_ * DH_;        // bf16 elements
  bf16* Qb = (bf16*)d_ws;
  bf16* Kb = Qb + perMat;
  bf16* Vb = Kb + perMat;

  dim3 g1(S_ / 128, B_ * H_, 3);
  qkv_proj_kernel<<<g1, dim3(256), 0, stream>>>(X, Wq, Wk, Wv, Qb, Kb, Vb);

  dim3 g2(S_ / 64, B_ * H_);
  flash_attn_kernel<<<g2, dim3(128), 0, stream>>>(Qb, Kb, Vb, (float*)d_out);
}
